// DSNN_67104569032984
// MI455X (gfx1250) — matrix operand order repaired and checked by reading, not compiled, not run on hardware
//
#include <hip/hip_runtime.h>

typedef __attribute__((ext_vector_type(16))) __bf16 v16bf;
typedef __attribute__((ext_vector_type(8)))  __bf16 v8bf;
typedef __attribute__((ext_vector_type(8)))  float  v8f;
typedef __attribute__((ext_vector_type(4)))  float  v4f;

#define BATCH  64
#define TSTEPS 1000
#define DINDIM 512
#define HDIM   512
#define MROWS  (BATCH * TSTEPS)

#define ALPHA_C 0.9f
#define BETA_C  0.85f
#define THR_C   0.1f

// ---------------------------------------------------------------------------
// Pass 0: split f32 -> bf16 (hi) + bf16 (residual lo), round-to-nearest-even,
// 8 elements per thread. Pure streaming; src is never needed as f32 again.
// ---------------------------------------------------------------------------
__global__ void __launch_bounds__(256)
split_to_bf16(const float* __restrict__ src,
              __bf16* __restrict__ hi, __bf16* __restrict__ lo) {
  const size_t base = ((size_t)blockIdx.x * blockDim.x + threadIdx.x) * 8;
  const v4f* p = reinterpret_cast<const v4f*>(src + base);
  v4f x0 = __builtin_nontemporal_load(p);
  v4f x1 = __builtin_nontemporal_load(p + 1);
  float xs[8] = {x0.x, x0.y, x0.z, x0.w, x1.x, x1.y, x1.z, x1.w};
  v8bf h, l;
#pragma unroll
  for (int i = 0; i < 8; ++i) {
    float x = xs[i];
    unsigned u  = __float_as_uint(x);
    unsigned rh = (u + 0x7FFFu + ((u >> 16) & 1u)) >> 16;
    h[i] = __builtin_bit_cast(__bf16, (unsigned short)rh);
    float xl = x - __uint_as_float(rh << 16);
    unsigned ul = __float_as_uint(xl);
    unsigned rl = (ul + 0x7FFFu + ((ul >> 16) & 1u)) >> 16;
    l[i] = __builtin_bit_cast(__bf16, (unsigned short)rl);
  }
  *reinterpret_cast<v8bf*>(hi + base) = h;
  *reinterpret_cast<v8bf*>(lo + base) = l;
}

// ---------------------------------------------------------------------------
// Pass 1: hs[m, n] = sum_k A[m, k] * W[n, k], bf16x3 (hi*hi + hi*lo + lo*hi),
// fp32 accumulate. One block = one 16-row M tile; each of 8 waves owns a
// 64-wide N strip (4 accumulators). Operands pre-split into packed bf16, so
// the hot loop is loads + v_wmma only.
// ---------------------------------------------------------------------------
__global__ void __launch_bounds__(256)
snn_gemm_bf16x3(const __bf16* __restrict__ Ahi, const __bf16* __restrict__ Alo,
                const __bf16* __restrict__ Whi, const __bf16* __restrict__ Wlo,
                float* __restrict__ hs) {
  const int lane = threadIdx.x & 31;
  const int wave = threadIdx.x >> 5;   // 0..7 -> N strip of 64
  const int half = lane >> 4;          // half-wave selector (ISA 7.12.2 layout)
  const int l    = lane & 15;
  const int m0   = blockIdx.x * 16;
  const int n0   = wave * 64;

  const size_t arow = (size_t)(m0 + l) * DINDIM;

  v8f acc[4];
#pragma unroll
  for (int j = 0; j < 4; ++j) acc[j] = (v8f){0.f,0.f,0.f,0.f,0.f,0.f,0.f,0.f};

#pragma unroll 2
  for (int kk = 0; kk < DINDIM; kk += 32) {
    // A tile 16x32: lane(l,half) holds row m0+l,
    // K = kk + half*16 + {0..15}  -> two 16B chunks per operand
    const size_t ka = arow + kk + half * 16;
    v8bf ah0 = *reinterpret_cast<const v8bf*>(Ahi + ka);
    v8bf ah1 = *reinterpret_cast<const v8bf*>(Ahi + ka + 8);
    v8bf al0 = *reinterpret_cast<const v8bf*>(Alo + ka);
    v8bf al1 = *reinterpret_cast<const v8bf*>(Alo + ka + 8);
    v16bf ahi = __builtin_shufflevector(ah0, ah1, 0,1,2,3,4,5,6,7,8,9,10,11,12,13,14,15);
    v16bf alo = __builtin_shufflevector(al0, al1, 0,1,2,3,4,5,6,7,8,9,10,11,12,13,14,15);

    // B tile 32x16 (B[k][n] = W[n][k]): lane(l,half) holds column n0+j*16+l,
    // K = kk + half*16 + {0..15}  -> 32 contiguous bytes of a W row
    const int kb = kk + half * 16;
#pragma unroll
    for (int j = 0; j < 4; ++j) {
      const size_t wrow = (size_t)(n0 + j * 16 + l) * DINDIM + kb;
      v16bf bhi = *reinterpret_cast<const v16bf*>(Whi + wrow);
      v16bf blo = *reinterpret_cast<const v16bf*>(Wlo + wrow);

      acc[j] = __builtin_amdgcn_wmma_f32_16x16x32_bf16(
                   false, ahi, false, bhi, (short)0, acc[j], false, false);
      acc[j] = __builtin_amdgcn_wmma_f32_16x16x32_bf16(
                   false, ahi, false, blo, (short)0, acc[j], false, false);
      acc[j] = __builtin_amdgcn_wmma_f32_16x16x32_bf16(
                   false, alo, false, bhi, (short)0, acc[j], false, false);
    }
  }

  // Store C: acc[j][r] is element (M = m0 + r + half*8, N = n0 + j*16 + l)
#pragma unroll
  for (int j = 0; j < 4; ++j) {
    const int nc = n0 + j * 16 + l;
#pragma unroll
    for (int r = 0; r < 8; ++r) {
      const size_t row = (size_t)(m0 + r + half * 8);
      hs[row * HDIM + nc] = acc[j][r];
    }
  }
}

// ---------------------------------------------------------------------------
// Pass 2: sequential LIF scan over T. One thread per (b, h). hs was staged in
// the syn_out region; each element is read once then overwritten with new_syn.
// ---------------------------------------------------------------------------
__global__ void __launch_bounds__(256)
snn_scan(float* __restrict__ out) {
  const size_t BTH = (size_t)BATCH * TSTEPS * HDIM;
  const size_t BH  = (size_t)BATCH * HDIM;
  float* spk  = out;
  float* memf = out + BTH;
  float* synf = out + BTH + BH;
  float* memo = out + BTH + 2 * BH;
  float* syno = out + 2 * BTH + 2 * BH;   // holds hs on entry

  const int i = blockIdx.x * blockDim.x + threadIdx.x;   // 0 .. B*H-1
  const int b = i / HDIM;
  const int h = i % HDIM;
  size_t idx = (size_t)b * TSTEPS * HDIM + h;

  float mem = 0.f, syn = 0.f;
  for (int t = 0; t < TSTEPS; ++t, idx += HDIM) {
    float hv = syno[idx];
    syn = ALPHA_C * syn + hv;
    mem = BETA_C * mem + syn;
    float mthr = mem - THR_C;
    float s = (mthr > 0.f) ? 1.f : 0.f;
    mem = (mthr > 0.f) ? 0.f : mem;
    __builtin_nontemporal_store(s,   spk  + idx);   // streamed, never re-read
    __builtin_nontemporal_store(mem, memo + idx);   // streamed, never re-read
    syno[idx] = syn;                                // in-place over hs
  }
  memf[i] = mem;
  synf[i] = syn;
}

extern "C" void kernel_launch(void* const* d_in, const int* in_sizes, int n_in,
                              void* d_out, int out_size, void* d_ws, size_t ws_size,
                              hipStream_t stream) {
  (void)in_sizes; (void)n_in; (void)out_size; (void)d_ws; (void)ws_size;
  const float* inputs  = (const float*)d_in[0];   // [B, T, DIN] f32
  const float* weights = (const float*)d_in[1];   // [H, DIN]    f32
  float* out = (float*)d_out;

  const size_t BTH = (size_t)BATCH * TSTEPS * HDIM;        // 32,768,000
  const size_t BH  = (size_t)BATCH * HDIM;                 // 32,768
  const size_t WN  = (size_t)HDIM * DINDIM;                // 262,144

  // Staging plan (all inside d_out, all regions rewritten by later passes):
  //   spk region     : W_hi | W_lo   (1 MB)   -- consumed by GEMM, then scan writes spk
  //   mem_out region : A_hi | A_lo  (131 MB)  -- consumed by GEMM, then scan writes mem_out
  //   syn_out region : hs           (131 MB)  -- scan reads + overwrites in place
  __bf16* Whi = (__bf16*)out;
  __bf16* Wlo = Whi + WN;
  __bf16* Ahi = (__bf16*)(out + BTH + 2 * BH);
  __bf16* Alo = Ahi + BTH;
  float*  hs  = out + 2 * BTH + 2 * BH;

  // Pass 0: split A (32768000 el = 256*8*16000) and W (262144 el = 256*8*128)
  split_to_bf16<<<dim3((unsigned)(BTH / (256 * 8))), dim3(256), 0, stream>>>(inputs, Ahi, Alo);
  split_to_bf16<<<dim3((unsigned)(WN  / (256 * 8))), dim3(256), 0, stream>>>(weights, Whi, Wlo);

  // Pass 1: GEMM (bf16x3 WMMA)
  snn_gemm_bf16x3<<<dim3(MROWS / 16), dim3(256), 0, stream>>>(Ahi, Alo, Whi, Wlo, hs);

  // Pass 2: LIF scan
  snn_scan<<<dim3((unsigned)(BH / 256)), dim3(256), 0, stream>>>(out);
}
